// RNN_8847632630090
// MI455X (gfx1250) — compile-verified, matrix operand order checked by reading, not run on hardware
//
#include <hip/hip_runtime.h>
#include <hip/hip_bf16.h>

typedef __attribute__((ext_vector_type(16))) _Float16 v16h;
typedef __attribute__((ext_vector_type(8)))  float    v8f;

#define HID   32
#define BATCH 256
#define SEQ   8192

// Native v_tanh_f32 if the toolchain exposes it for gfx1250; otherwise
// tanh(v) = 1 - 2/(exp(2v)+1) via single-cycle-issue transcendentals.
__device__ __forceinline__ float fast_tanh(float v) {
#if __has_builtin(__builtin_amdgcn_tanhf)
    return __builtin_amdgcn_tanhf(v);
#elif __has_builtin(__builtin_amdgcn_tanh_f32)
    return __builtin_amdgcn_tanh_f32(v);
#else
    float e = __builtin_amdgcn_exp2f(v * 2.8853900817779268f);  // 2*log2(e)
    float r = __builtin_amdgcn_rcpf(e + 1.0f);
    return fmaf(-2.0f, r, 1.0f);
#endif
}

// One wave (32 lanes) per block handles a 16-row batch tile across all 8192 steps.
// h lives in registers in WMMA B-operand layout; W_hh lives in registers in the
// A-operand layout (two 16x32 tiles). D-layout of the step output aligns lane-wise
// with the next step's B-layout, so the recurrence never leaves the wave.
__global__ __launch_bounds__(32) void rnn_wmma_kernel(
    const float* __restrict__ x,      // [B,S]
    const float* __restrict__ h0,     // [B,32]
    const float* __restrict__ w_ih,   // [32]
    const float* __restrict__ w_hh,   // [32,32] row-major (w_hh[j][k])
    const float* __restrict__ b_ih,   // [32]
    const float* __restrict__ b_hh,   // [32]
    const float* __restrict__ w_out,  // [32]
    const float* __restrict__ b_out,  // [1]
    float* __restrict__ y,            // [B,S]
    float* __restrict__ h_last)       // [B,32]
{
    const int lane = threadIdx.x & 31;
    const int half = lane >> 4;       // 0: lanes 0-15, 1: lanes 16-31
    const int bl   = lane & 15;
    const int b    = blockIdx.x * 16 + bl;

    // ---- A tiles: w_hh in 16-bit A layout (element e: e<8 -> K=8*half+e,
    //      e>=8 -> K=16+8*half+(e-8)); tile0: M=j=bl, tile1: M=j=16+bl ----
    v16h a0{}, a1{};
#pragma unroll
    for (int e = 0; e < 16; ++e) {
        int k = (e < 8) ? (8 * half + e) : (16 + 8 * half + (e - 8));
        a0[e] = (_Float16)w_hh[bl * HID + k];
        a1[e] = (_Float16)w_hh[(16 + bl) * HID + k];
    }

    // ---- Per-lane constants for the C operand and the output projection.
    //      C/D layout: this lane owns (j = 8*half + r) for tile0,
    //      (j = 16 + 8*half + r) for tile1, r = VGPR index 0..7, column b = bl.
    float wih0[8], wih1[8], bias0[8], bias1[8], wo0[8], wo1[8];
#pragma unroll
    for (int r = 0; r < 8; ++r) {
        int j0 = 8 * half + r;
        int j1 = 16 + 8 * half + r;
        wih0[r]  = w_ih[j0];             wih1[r]  = w_ih[j1];
        bias0[r] = b_ih[j0] + b_hh[j0];  bias1[r] = b_ih[j1] + b_hh[j1];
        wo0[r]   = w_out[j0];            wo1[r]   = w_out[j1];
    }
    const float bo = b_out[0];

    // ---- Initial h in B-operand layout (element e -> K=j as for A layout) ----
    v16h hB{};
#pragma unroll
    for (int e = 0; e < 16; ++e) {
        int j = (e < 8) ? (8 * half + e) : (16 + 8 * half + (e - 8));
        hB[e] = (_Float16)h0[b * HID + j];
    }

    const float4* __restrict__ x4 = reinterpret_cast<const float4*>(x + (size_t)b * SEQ);
    float* __restrict__ yb = y + (size_t)b * SEQ;

#pragma unroll 1
    for (int s4 = 0; s4 < SEQ / 4; ++s4) {
        float4 xv = x4[s4];
        float xs[4] = {xv.x, xv.y, xv.z, xv.w};
#pragma unroll
        for (int u = 0; u < 4; ++u) {
            int s = s4 * 4 + u;
            // C operand: x_t * w_ih[j] + (b_ih[j] + b_hh[j])
            v8f c0, c1;
#pragma unroll
            for (int r = 0; r < 8; ++r) {
                c0[r] = fmaf(xs[u], wih0[r], bias0[r]);
                c1[r] = fmaf(xs[u], wih1[r], bias1[r]);
            }
            // D[j,b] = sum_k w_hh[j,k] * h[k,b] + C   (two j-tiles, shared B=h)
            v8f d0 = __builtin_amdgcn_wmma_f32_16x16x32_f16(
                false, a0, false, hB, (short)0, c0, false, false);
            v8f d1 = __builtin_amdgcn_wmma_f32_16x16x32_f16(
                false, a1, false, hB, (short)0, c1, false, false);
            // tanh, repack into next-step B operand (lane-local!), partial y dot
            float p = 0.0f;
#pragma unroll
            for (int r = 0; r < 8; ++r) {
                float t0 = fast_tanh(d0[r]);
                float t1 = fast_tanh(d1[r]);
                hB[r]     = (_Float16)t0;
                hB[8 + r] = (_Float16)t1;
                p = fmaf(t0, wo0[r], p);
                p = fmaf(t1, wo1[r], p);
            }
            // combine lane halves (j-lower half in lanes 0-15, upper in 16-31):
            // ds_swizzle SWAPX16: src_lane = lane ^ 16
            int pi = __float_as_int(p);
            float po = __int_as_float(__builtin_amdgcn_ds_swizzle(pi, 0x401F));
            float yv = p + po + bo;
            yb[s] = yv;   // both halves write the same value: benign, keeps EXEC full
        }
    }

    // ---- final hidden state ----
#pragma unroll
    for (int e = 0; e < 16; ++e) {
        int j = (e < 8) ? (8 * half + e) : (16 + 8 * half + (e - 8));
        h_last[b * HID + j] = (float)hB[e];
    }
}

extern "C" void kernel_launch(void* const* d_in, const int* in_sizes, int n_in,
                              void* d_out, int out_size, void* d_ws, size_t ws_size,
                              hipStream_t stream) {
    const float* x     = (const float*)d_in[0];
    const float* h0    = (const float*)d_in[1];
    const float* w_ih  = (const float*)d_in[2];
    const float* w_hh  = (const float*)d_in[3];
    const float* b_ih  = (const float*)d_in[4];
    const float* b_hh  = (const float*)d_in[5];
    const float* w_out = (const float*)d_in[6];
    const float* b_out = (const float*)d_in[7];

    float* y      = (float*)d_out;                       // [B,S,1] flat
    float* h_last = (float*)d_out + (size_t)BATCH * SEQ; // [1,B,H] flat

    rnn_wmma_kernel<<<dim3(BATCH / 16), dim3(32), 0, stream>>>(
        x, h0, w_ih, w_hh, b_ih, b_hh, w_out, b_out, y, h_last);
}
